// MoCTopKExperts_72627896976025
// MI455X (gfx1250) — matrix operand order repaired and checked by reading, not compiled, not run on hardware
//
#include <hip/hip_runtime.h>
#include <hip/hip_bf16.h>
#include <math.h>
#include <stdint.h>

// ---------------- problem constants (from reference) ----------------
#define Dd 1024
#define Ee 8
#define Hh 2816
#define Ntok 4096        // B*T
#define NKtot 8192       // N*K
#define CAPc 1280        // ceil(NK/E * 1.25)
#define ECAP 10240       // E*CAP
#define OUT_FUSED 4194304  // N*D

typedef __attribute__((ext_vector_type(16))) __bf16 v16bf;
typedef __attribute__((ext_vector_type(8)))  float v8f;
typedef __attribute__((ext_vector_type(4)))  unsigned int v4u;

union Frag { v16bf v; v4u q[2]; unsigned int u[8]; };

#define LDSTRIDE 40   // halves per LDS tile row: 80B, 16B-aligned, conflict-free

__device__ __forceinline__ unsigned short f2bf(float f) {
  unsigned int u = __float_as_uint(f);
  unsigned int r = u + 0x7FFFu + ((u >> 16) & 1u);   // RNE
  return (unsigned short)(r >> 16);
}

__device__ __forceinline__ void async_b128_to_lds(unsigned ldsOff, const void* gaddr) {
  // CDNA5 async DMA: global -> LDS, tracked by ASYNCcnt
  asm volatile("global_load_async_to_lds_b128 %0, %1, off"
               :: "v"(ldsOff), "v"((unsigned long long)(uintptr_t)gaddr) : "memory");
}
__device__ __forceinline__ void wait_async0() {
  asm volatile("s_wait_asynccnt 0" ::: "memory");
}

// =====================================================================
// Generic bf16 GEMM: C[M,N] = A[M,K](bf16,row) * B[K,N](f32,row -> bf16)
// block tile 128x64x32, 256 threads = 8 waves, wave tile 32x32 (2x2 WMMA)
// Double-buffered LDS; A tile copied via global_load_async_to_lds_b128,
// B tile prefetched to regs one stage ahead, converted to bf16, ds_stored.
// epi: bit0 = store f32 to Cf, bit1 = store bf16 to Cb, bit2 = GELU first
// =====================================================================
__global__ __launch_bounds__(256)
void k_gemm(const unsigned short* __restrict__ A, long sA, int lda,
            const float* __restrict__ B, long sB, int ldb,
            float* __restrict__ Cf, unsigned short* __restrict__ Cb,
            long sC, int ldc, int Kdim, int epi)
{
  __shared__ __align__(16) unsigned short As[2][128 * LDSTRIDE];
  __shared__ __align__(16) unsigned short Bs[2][64 * LDSTRIDE];
  const int tid = threadIdx.x;
  const int lane = tid & 31, w = tid >> 5;
  const int wm = w & 3, wn = w >> 2;
  const int m0 = blockIdx.y * 128, n0 = blockIdx.x * 64;
  const int r16 = lane & 15, hh = lane >> 4;
  A += (long)blockIdx.z * sA;
  B += (long)blockIdx.z * sB;
  v8f acc[2][2] = {};
  float rB[8];

  auto loadB = [&](int kb) {
    #pragma unroll
    for (int qq = 0; qq < 4; ++qq) {
      int li = qq * 256 + tid;           // pair id 0..1023
      int c = li & 63, kp = li >> 6;     // kp 0..15
      rB[2 * qq]     = B[(long)(kb + 2 * kp) * ldb + n0 + c];
      rB[2 * qq + 1] = B[(long)(kb + 2 * kp + 1) * ldb + n0 + c];
    }
  };
  auto storeB = [&](int bu) {
    #pragma unroll
    for (int qq = 0; qq < 4; ++qq) {
      int li = qq * 256 + tid;
      int c = li & 63, kp = li >> 6;
      unsigned pk = (unsigned)f2bf(rB[2 * qq]) | ((unsigned)f2bf(rB[2 * qq + 1]) << 16);
      *(unsigned int*)(&Bs[bu][c * LDSTRIDE + 2 * kp]) = pk;
    }
  };
  auto asyncA = [&](int kb, int bu) {
    #pragma unroll
    for (int qq = 0; qq < 2; ++qq) {
      int li = tid * 2 + qq;                 // 0..511 b128 chunks
      int r = li >> 2, seg = (li & 3) * 8;   // seg in halves
      unsigned lo = (unsigned)(uintptr_t)(&As[bu][r * LDSTRIDE + seg]);
      async_b128_to_lds(lo, A + (long)(m0 + r) * lda + kb + seg);
    }
  };

  loadB(0);
  asyncA(0, 0);
  int buf = 0;
  for (int kb = 0; kb < Kdim; kb += 32, buf ^= 1) {
    storeB(buf);
    wait_async0();
    __syncthreads();
    if (kb + 32 < Kdim) { asyncA(kb + 32, buf ^ 1); loadB(kb + 32); }
    Frag af[2], bf2[2];
    #pragma unroll
    for (int mt = 0; mt < 2; ++mt) {
      int rr = wm * 32 + mt * 16 + r16;
      af[mt].q[0] = *(const v4u*)(&As[buf][rr * LDSTRIDE + hh * 8]);
      af[mt].q[1] = *(const v4u*)(&As[buf][rr * LDSTRIDE + hh * 8 + 16]);
    }
    #pragma unroll
    for (int nt = 0; nt < 2; ++nt) {
      int cc = wn * 32 + nt * 16 + r16;
      bf2[nt].q[0] = *(const v4u*)(&Bs[buf][cc * LDSTRIDE + hh * 8]);
      bf2[nt].q[1] = *(const v4u*)(&Bs[buf][cc * LDSTRIDE + hh * 8 + 16]);
    }
    #pragma unroll
    for (int mt = 0; mt < 2; ++mt)
      #pragma unroll
      for (int nt = 0; nt < 2; ++nt)
        acc[mt][nt] = __builtin_amdgcn_wmma_f32_16x16x32_bf16(
            false, af[mt].v, false, bf2[nt].v, (short)0, acc[mt][nt], false, false);
  }
  #pragma unroll
  for (int mt = 0; mt < 2; ++mt)
    #pragma unroll
    for (int nt = 0; nt < 2; ++nt) {
      int col = n0 + wn * 32 + nt * 16 + r16;
      #pragma unroll
      for (int v = 0; v < 8; ++v) {
        int row = m0 + wm * 32 + mt * 16 + v + hh * 8;
        float val = acc[mt][nt][v];
        if (epi & 4) val = 0.5f * val * (1.0f + erff(val * 0.70710678118654752f));
        long o = (long)blockIdx.z * sC + (long)row * ldc + col;
        if (epi & 1) Cf[o] = val;
        if (epi & 2) Cb[o] = f2bf(val);
      }
    }
}

// =====================================================================
// Fused SwiGLU GEMM: act = silu(A*Wg) * (A*Wu), Wg = w13[:, n], Wu = w13[:, H+n]
// A: [CAP x 1024] bf16 (batched over E), B: [1024 x 5632] f32, Act: [CAP x 2816] bf16
// =====================================================================
__global__ __launch_bounds__(256)
void k_gemm_swiglu(const unsigned short* __restrict__ A, long sA,
                   const float* __restrict__ B, long sB,
                   unsigned short* __restrict__ Act, long sAct)
{
  __shared__ __align__(16) unsigned short As[2][128 * LDSTRIDE];
  __shared__ __align__(16) unsigned short Bs[2][2][64 * LDSTRIDE];
  const int tid = threadIdx.x;
  const int lane = tid & 31, w = tid >> 5;
  const int wm = w & 3, wn = w >> 2;
  const int m0 = blockIdx.y * 128, n0 = blockIdx.x * 64;
  const int r16 = lane & 15, hh = lane >> 4;
  A += (long)blockIdx.z * sA;
  B += (long)blockIdx.z * sB;
  v8f ag[2][2] = {}, au[2][2] = {};
  float rB[2][8];

  auto loadB = [&](int kb) {
    #pragma unroll
    for (int hv = 0; hv < 2; ++hv)
      #pragma unroll
      for (int qq = 0; qq < 4; ++qq) {
        int li = qq * 256 + tid;
        int c = li & 63, kp = li >> 6;
        rB[hv][2 * qq]     = B[(long)(kb + 2 * kp) * 5632 + n0 + hv * Hh + c];
        rB[hv][2 * qq + 1] = B[(long)(kb + 2 * kp + 1) * 5632 + n0 + hv * Hh + c];
      }
  };
  auto storeB = [&](int bu) {
    #pragma unroll
    for (int hv = 0; hv < 2; ++hv)
      #pragma unroll
      for (int qq = 0; qq < 4; ++qq) {
        int li = qq * 256 + tid;
        int c = li & 63, kp = li >> 6;
        unsigned pk = (unsigned)f2bf(rB[hv][2 * qq]) | ((unsigned)f2bf(rB[hv][2 * qq + 1]) << 16);
        *(unsigned int*)(&Bs[bu][hv][c * LDSTRIDE + 2 * kp]) = pk;
      }
  };
  auto asyncA = [&](int kb, int bu) {
    #pragma unroll
    for (int qq = 0; qq < 2; ++qq) {
      int li = tid * 2 + qq;
      int r = li >> 2, seg = (li & 3) * 8;
      unsigned lo = (unsigned)(uintptr_t)(&As[bu][r * LDSTRIDE + seg]);
      async_b128_to_lds(lo, A + (long)(m0 + r) * 1024 + kb + seg);
    }
  };

  loadB(0);
  asyncA(0, 0);
  int buf = 0;
  for (int kb = 0; kb < 1024; kb += 32, buf ^= 1) {
    storeB(buf);
    wait_async0();
    __syncthreads();
    if (kb + 32 < 1024) { asyncA(kb + 32, buf ^ 1); loadB(kb + 32); }
    Frag af[2], bg[2], bu2[2];
    #pragma unroll
    for (int mt = 0; mt < 2; ++mt) {
      int rr = wm * 32 + mt * 16 + r16;
      af[mt].q[0] = *(const v4u*)(&As[buf][rr * LDSTRIDE + hh * 8]);
      af[mt].q[1] = *(const v4u*)(&As[buf][rr * LDSTRIDE + hh * 8 + 16]);
    }
    #pragma unroll
    for (int nt = 0; nt < 2; ++nt) {
      int cc = wn * 32 + nt * 16 + r16;
      bg[nt].q[0]  = *(const v4u*)(&Bs[buf][0][cc * LDSTRIDE + hh * 8]);
      bg[nt].q[1]  = *(const v4u*)(&Bs[buf][0][cc * LDSTRIDE + hh * 8 + 16]);
      bu2[nt].q[0] = *(const v4u*)(&Bs[buf][1][cc * LDSTRIDE + hh * 8]);
      bu2[nt].q[1] = *(const v4u*)(&Bs[buf][1][cc * LDSTRIDE + hh * 8 + 16]);
    }
    #pragma unroll
    for (int mt = 0; mt < 2; ++mt)
      #pragma unroll
      for (int nt = 0; nt < 2; ++nt) {
        ag[mt][nt] = __builtin_amdgcn_wmma_f32_16x16x32_bf16(
            false, af[mt].v, false, bg[nt].v, (short)0, ag[mt][nt], false, false);
        au[mt][nt] = __builtin_amdgcn_wmma_f32_16x16x32_bf16(
            false, af[mt].v, false, bu2[nt].v, (short)0, au[mt][nt], false, false);
      }
  }
  #pragma unroll
  for (int mt = 0; mt < 2; ++mt)
    #pragma unroll
    for (int nt = 0; nt < 2; ++nt) {
      int col = n0 + wn * 32 + nt * 16 + r16;
      #pragma unroll
      for (int v = 0; v < 8; ++v) {
        int row = m0 + wm * 32 + mt * 16 + v + hh * 8;
        float g = ag[mt][nt][v], u = au[mt][nt][v];
        float s = g / (1.0f + expf(-g));          // silu
        Act[(long)blockIdx.z * sAct + (long)row * Hh + col] = f2bf(s * u);
      }
    }
}

// ---------------- small kernels ----------------
__global__ void k_init(float* accum, int* row_src) {
  int i = blockIdx.x * 256 + threadIdx.x;
  if (i < 32) accum[i] = 0.0f;
  if (i < ECAP) row_src[i] = -1;
}

__global__ __launch_bounds__(256)
void k_router(const float* __restrict__ x, const float* __restrict__ gw,
              int* __restrict__ tgt, float* __restrict__ probs,
              float* __restrict__ accum, int* __restrict__ outIdx)
{
  __shared__ float g[1024 * 8];
  int tid = threadIdx.x;
  for (int i = tid; i < 8192; i += 256) g[i] = gw[i];
  __syncthreads();
  int n = blockIdx.x * 256 + tid;
  const float* xr = x + (long)n * 1024;
  float lg[8] = {0, 0, 0, 0, 0, 0, 0, 0};
  for (int d = 0; d < 1024; ++d) {
    float xv = xr[d];
    #pragma unroll
    for (int e = 0; e < 8; ++e) lg[e] = fmaf(xv, g[d * 8 + e], lg[e]);
  }
  int i0 = 0; float v0 = lg[0];
  #pragma unroll
  for (int e = 1; e < 8; ++e) if (lg[e] > v0) { v0 = lg[e]; i0 = e; }
  int i1 = -1; float v1 = -3.4e38f;
  #pragma unroll
  for (int e = 0; e < 8; ++e) if (e != i0 && lg[e] > v1) { v1 = lg[e]; i1 = e; }
  float e1 = expf(v1 - v0);
  float inv2 = 1.0f / (1.0f + e1);
  tgt[2 * n] = i0;  tgt[2 * n + 1] = i1;
  probs[2 * n] = inv2;  probs[2 * n + 1] = e1 * inv2;
  outIdx[2 * n] = i0;  outIdx[2 * n + 1] = i1;
  float s = 0.0f;
  #pragma unroll
  for (int e = 0; e < 8; ++e) s += expf(lg[e] - v0);
  float invs = 1.0f / s;
  const float invN = 1.0f / (float)Ntok;
  #pragma unroll
  for (int e = 0; e < 8; ++e) atomicAdd(&accum[e], expf(lg[e] - v0) * invs * invN);
  atomicAdd(&accum[8 + i0], invN);
  float z = v0 + logf(s);
  atomicAdd(&accum[16], z * z * invN);
}

__global__ void k_aux(const float* accum, float* outAux) {
  if (threadIdx.x == 0 && blockIdx.x == 0) {
    float bal = 0.0f;
    for (int e = 0; e < 8; ++e) bal += accum[e] * accum[8 + e];
    outAux[0] = 0.01f * (bal * 8.0f) + 0.001f * accum[16];
  }
}

// deterministic stable rank-within-expert (matches jnp.argsort stability)
__global__ __launch_bounds__(256)
void k_rank(const int* __restrict__ tgt, int* __restrict__ keepArr,
            int* __restrict__ slotOf, int* __restrict__ row_src)
{
  __shared__ int sc[256];
  int e = blockIdx.x, t = threadIdx.x, running = 0;
  for (int base = 0; base < NKtot; base += 256) {
    int i = base + t;
    int flag = (tgt[i] == e) ? 1 : 0;
    sc[t] = flag;
    __syncthreads();
    for (int off = 1; off < 256; off <<= 1) {
      int v = (t >= off) ? sc[t - off] : 0;
      __syncthreads();
      sc[t] += v;
      __syncthreads();
    }
    int excl = sc[t] - flag;
    int total = sc[255];
    if (flag) {
      int rank = running + excl;
      int kept = rank < CAPc;
      keepArr[i] = kept;
      slotOf[i] = kept ? (e * CAPc + rank) : -1;
      if (kept) row_src[e * CAPc + rank] = i;
    }
    running += total;
    __syncthreads();
  }
}

__global__ __launch_bounds__(256)
void k_scatter(const float* __restrict__ x, const int* __restrict__ row_src,
               unsigned short* __restrict__ bufA)
{
  long idx = (long)blockIdx.x * 256 + threadIdx.x;
  int s = (int)(idx >> 10), d = (int)(idx & 1023);
  int src = row_src[s];
  float v = 0.0f;
  if (src >= 0) v = x[(long)(src >> 1) * 1024 + d];
  bufA[idx] = f2bf(v);
}

__global__ __launch_bounds__(256)
void k_combine(const float* __restrict__ y, const int* __restrict__ keep,
               const int* __restrict__ slotOf,
               float* __restrict__ self32, unsigned short* __restrict__ selbf)
{
  long idx = (long)blockIdx.x * 256 + threadIdx.x;
  int nk = (int)(idx >> 10), d = (int)(idx & 1023);
  float v = 0.0f;
  if (keep[nk]) v = y[(long)slotOf[nk] * 1024 + d];
  self32[idx] = v;
  selbf[idx] = f2bf(v);
}

__global__ __launch_bounds__(128)
void k_scores(const float* __restrict__ Q, const float* __restrict__ Kmat,
              const int* __restrict__ keep, float* __restrict__ Amat)
{
  int t = threadIdx.x, lane = t & 31, wv = t >> 5;
  int n = blockIdx.x * 4 + wv;
  const float* q0 = Q + (long)(2 * n) * 1024;
  const float* q1 = q0 + 1024;
  const float* k0 = Kmat + (long)(2 * n) * 1024;
  const float* k1 = k0 + 1024;
  float s00 = 0, s01 = 0, s10 = 0, s11 = 0;
  for (int d = lane; d < 1024; d += 32) {
    float a0 = q0[d], a1 = q1[d], b0 = k0[d], b1 = k1[d];
    s00 = fmaf(a0, b0, s00); s01 = fmaf(a0, b1, s01);
    s10 = fmaf(a1, b0, s10); s11 = fmaf(a1, b1, s11);
  }
  #pragma unroll
  for (int off = 16; off > 0; off >>= 1) {
    s00 += __shfl_xor(s00, off, 32);
    s01 += __shfl_xor(s01, off, 32);
    s10 += __shfl_xor(s10, off, 32);
    s11 += __shfl_xor(s11, off, 32);
  }
  if (lane == 0) {
    int c0 = keep[2 * n], c1 = keep[2 * n + 1];
    const float scl = 1.0f / 32.0f;   // 1/sqrt(1024)
    float m[2][2];
    int vd[2][2];
    vd[0][0] = c0 & c0; vd[0][1] = c0 & c1; vd[1][0] = c1 & c0; vd[1][1] = c1 & c1;
    m[0][0] = vd[0][0] ? s00 * scl : -1e9f;
    m[0][1] = vd[0][1] ? s01 * scl : -1e9f;
    m[1][0] = vd[1][0] ? s10 * scl : -1e9f;
    m[1][1] = vd[1][1] ? s11 * scl : -1e9f;
    #pragma unroll
    for (int i = 0; i < 2; ++i) {
      float mx = fmaxf(m[i][0], m[i][1]);
      float e0 = expf(m[i][0] - mx), ee1 = expf(m[i][1] - mx);
      float inv = 1.0f / (e0 + ee1);
      float p0 = e0 * inv * (float)vd[i][0];
      float p1 = ee1 * inv * (float)vd[i][1];
      float rinv = 1.0f / fmaxf(p0 + p1, 1e-12f);
      Amat[(2 * n + i) * 2 + 0] = p0 * rinv;
      Amat[(2 * n + i) * 2 + 1] = p1 * rinv;
    }
  }
}

__global__ __launch_bounds__(256)
void k_cmsg_cat(const float* __restrict__ self32, const float* __restrict__ Mf,
                const float* __restrict__ Amat, unsigned short* __restrict__ catb)
{
  long idx = (long)blockIdx.x * 256 + threadIdx.x;
  int nk = (int)(idx >> 10), d = (int)(idx & 1023);
  int n = nk >> 1;
  float a0 = Amat[nk * 2 + 0], a1 = Amat[nk * 2 + 1];
  float cm = a0 * Mf[(long)(2 * n) * 1024 + d] + a1 * Mf[(long)(2 * n + 1) * 1024 + d];
  catb[(long)nk * 2048 + d] = f2bf(self32[idx]);
  catb[(long)nk * 2048 + 1024 + d] = f2bf(cm);
}

__global__ __launch_bounds__(256)
void k_refine_fuse(const float* __restrict__ self32, const float* __restrict__ upd,
                   const int* __restrict__ keep, const float* __restrict__ probs,
                   unsigned short* __restrict__ fusedpre)
{
  long idx = (long)blockIdx.x * 256 + threadIdx.x;
  int n = (int)(idx >> 10), d = (int)(idx & 1023);
  long r0i = (long)(2 * n) * 1024 + d, r1i = r0i + 1024;
  float c0 = (float)keep[2 * n], c1 = (float)keep[2 * n + 1];
  float r0 = (self32[r0i] + upd[r0i]) * c0;
  float r1 = (self32[r1i] + upd[r1i]) * c1;
  float p0 = probs[2 * n] * c0, p1 = probs[2 * n + 1] * c1;
  float inv = 1.0f / fmaxf(p0 + p1, 1e-12f);
  fusedpre[(long)n * 1024 + d] = f2bf(r0 * (p0 * inv) + r1 * (p1 * inv));
}

// ---------------- host launcher ----------------
extern "C" void kernel_launch(void* const* d_in, const int* in_sizes, int n_in,
                              void* d_out, int out_size, void* d_ws, size_t ws_size,
                              hipStream_t stream) {
  (void)in_sizes; (void)n_in; (void)out_size; (void)ws_size;
  const float* x      = (const float*)d_in[0];
  const float* gate_w = (const float*)d_in[1];
  const float* w13    = (const float*)d_in[2];
  const float* w2     = (const float*)d_in[3];
  const float* msg_w  = (const float*)d_in[4];
  const float* q_w    = (const float*)d_in[5];
  const float* k_w    = (const float*)d_in[6];
  const float* upd_w1 = (const float*)d_in[7];
  const float* upd_w2 = (const float*)d_in[8];
  const float* o_w    = (const float*)d_in[9];
  float* out_f = (float*)d_out;
  int*   out_i = (int*)d_out;

  size_t off = 0;
  char* base = (char*)d_ws;
  auto alloc = [&](size_t bytes) -> char* {
    char* p = base + off;
    off += (bytes + 255) & ~(size_t)255;
    return p;
  };
  float*          accum   = (float*)alloc(32 * 4);
  int*            tgt     = (int*)alloc((size_t)NKtot * 4);
  int*            keepArr = (int*)alloc((size_t)NKtot * 4);
  int*            slotOf  = (int*)alloc((size_t)NKtot * 4);
  int*            row_src = (int*)alloc((size_t)ECAP * 4);
  float*          probs   = (float*)alloc((size_t)NKtot * 4);
  float*          Amat    = (float*)alloc((size_t)NKtot * 2 * 4);
  unsigned short* bufA    = (unsigned short*)alloc((size_t)ECAP * Dd * 2);
  unsigned short* act     = (unsigned short*)alloc((size_t)ECAP * Hh * 2);
  float*          y       = (float*)alloc((size_t)ECAP * Dd * 4);      // reused as upd
  float*          self32  = (float*)alloc((size_t)NKtot * Dd * 4);
  unsigned short* selbf   = (unsigned short*)alloc((size_t)NKtot * Dd * 2);
  float*          Mf      = (float*)alloc((size_t)NKtot * Dd * 4);
  unsigned short* Mbf     = (unsigned short*)alloc((size_t)NKtot * Dd * 2);
  float*          Qf      = (float*)alloc((size_t)NKtot * Dd * 4);     // reused as hid (bf16)
  float*          Kkf     = (float*)alloc((size_t)NKtot * Dd * 4);     // reused as cat (bf16)
  unsigned short* fusedpre= (unsigned short*)alloc((size_t)Ntok * Dd * 2);
  float*          upd     = y;
  unsigned short* hid     = (unsigned short*)Qf;
  unsigned short* catb    = (unsigned short*)Kkf;

  // 1) init + router + aux + rank + scatter
  k_init<<<40, 256, 0, stream>>>(accum, row_src);
  k_router<<<16, 256, 0, stream>>>(x, gate_w, tgt, probs, accum, out_i + (OUT_FUSED + 1));
  k_aux<<<1, 32, 0, stream>>>(accum, out_f + OUT_FUSED);
  k_rank<<<Ee, 256, 0, stream>>>(tgt, keepArr, slotOf, row_src);
  k_scatter<<<(ECAP * Dd) / 256, 256, 0, stream>>>(x, row_src, bufA);

  // 2) grouped SwiGLU FFN (bf16 WMMA)
  k_gemm_swiglu<<<dim3(Hh / 64, CAPc / 128, Ee), 256, 0, stream>>>(
      bufA, (long)CAPc * Dd, w13, (long)Dd * 2 * Hh, act, (long)CAPc * Hh);
  k_gemm<<<dim3(Dd / 64, CAPc / 128, Ee), 256, 0, stream>>>(
      act, (long)CAPc * Hh, Hh, w2, (long)Hh * Dd, Dd,
      y, nullptr, (long)CAPc * Dd, Dd, Hh, 1);

  // 3) combine back to token-slot order
  k_combine<<<(NKtot * Dd) / 256, 256, 0, stream>>>(y, keepArr, slotOf, self32, selbf);

  // 4) collaboration GEMMs
  k_gemm<<<dim3(Dd / 64, NKtot / 128, 1), 256, 0, stream>>>(
      selbf, 0, Dd, msg_w, 0, Dd, Mf, Mbf, 0, Dd, Dd, 3);          // M (f32 + bf16)
  k_gemm<<<dim3(Dd / 64, NKtot / 128, 1), 256, 0, stream>>>(
      selbf, 0, Dd, q_w, 0, Dd, Qf, nullptr, 0, Dd, Dd, 1);        // Q
  k_gemm<<<dim3(Dd / 64, NKtot / 128, 1), 256, 0, stream>>>(
      Mbf, 0, Dd, k_w, 0, Dd, Kkf, nullptr, 0, Dd, Dd, 1);         // Kk
  k_scores<<<Ntok / 4, 128, 0, stream>>>(Qf, Kkf, keepArr, Amat);
  k_cmsg_cat<<<(NKtot * Dd) / 256, 256, 0, stream>>>(self32, Mf, Amat, catb);
  k_gemm<<<dim3(2 * Dd / 64, NKtot / 128, 1), 256, 0, stream>>>(
      catb, 0, 2 * Dd, upd_w1, 0, 2 * Dd, nullptr, hid, 0, 2 * Dd, 2 * Dd, 6); // GELU->bf16
  k_gemm<<<dim3(Dd / 64, NKtot / 128, 1), 256, 0, stream>>>(
      hid, 0, 2 * Dd, upd_w2, 0, Dd, upd, nullptr, 0, Dd, 2 * Dd, 1);

  // 5) refine + weighted fusion + output projection
  k_refine_fuse<<<(Ntok * Dd) / 256, 256, 0, stream>>>(self32, upd, keepArr, probs, fusedpre);
  k_gemm<<<dim3(Dd / 64, Ntok / 128, 1), 256, 0, stream>>>(
      fusedpre, 0, Dd, o_w, 0, Dd, out_f, nullptr, 0, Dd, Dd, 1);
}
